// CrossAttention_20014547599675
// MI455X (gfx1250) — compile-verified, matrix operand order checked by reading, not run on hardware
//
#include <hip/hip_runtime.h>

typedef __attribute__((ext_vector_type(16))) _Float16 v16h;
typedef __attribute__((ext_vector_type(8)))  _Float16 v8h;
typedef __attribute__((ext_vector_type(8)))  float    v8f;
typedef __attribute__((ext_vector_type(4)))  unsigned int u32x4;
typedef __attribute__((ext_vector_type(8)))  int      i32x8;
typedef __attribute__((ext_vector_type(4)))  int      i32x4;

#define HEADS   8
#define DH      64
#define CCH     320     // channels
#define TTOK    16      // tokens per sequence
#define INNER   512     // HEADS*DH
#define KT_X    10      // 320/32  k-tiles for QKV gemm
#define KT_O    16      // 512/32  k-tiles for out-proj gemm
#define NT_QKV  96      // 1536/16 n-tiles (Q|K|V concatenated)
#define NT_OUT  20      // 320/16  n-tiles out-proj
#define SEQ_WG  2       // sequences per workgroup
#define ROWS    32      // SEQ_WG*16 token rows per workgroup
#define NSEQ    4096    // 4*32*32
#define HW      1024    // 32*32 spatial

#define QKV_PACK_HALVES (NT_QKV * KT_X * 512)
#define O_PACK_HALVES   (NT_OUT * KT_O * 512)

static __device__ __forceinline__ v8f wmma_f16(v16h a, v16h b, v8f c) {
    return __builtin_amdgcn_wmma_f32_16x16x32_f16(false, a, false, b, (short)0, c,
                                                  false, false);
}

// A/B 16-bit fragment index math (ISA 7.12.2):
//   lane = g*16 + m ; halves i<8 -> K = 8g+i ; i>=8 -> K = 16 + 8g + (i-8)
static __device__ __forceinline__ int frag_i_from_kk(int kk) {
    return (kk & 7) + ((kk & 16) ? 8 : 0);
}
static __device__ __forceinline__ int frag_g_from_kk(int kk) {
    return (kk >> 3) & 1;
}

// ---------------------------------------------------------------------------
// Kernel 0: repack fp32 weights into f16 WMMA B-fragment layout in d_ws.
//   [0, QKV_PACK_HALVES)              : [Wq;Wk;Wv]^T tiles, ((nt*KT_X+ko)*32+lane)*16+i
//   [QKV_PACK_HALVES, +O_PACK_HALVES) : Wo^T tiles,         ((co*KT_O+ko)*32+lane)*16+i
// ---------------------------------------------------------------------------
__global__ void ca_pack_weights(const float* __restrict__ Wq,
                                const float* __restrict__ Wk,
                                const float* __restrict__ Wv,
                                const float* __restrict__ Wo,
                                _Float16* __restrict__ pk) {
    int idx = blockIdx.x * 256 + threadIdx.x;
    if (idx < QKV_PACK_HALVES) {
        int tile = idx >> 9, rem = idx & 511;
        int lane = rem >> 4, i = rem & 15;
        int nt = tile / KT_X, ko = tile % KT_X;
        int g = lane >> 4, n = lane & 15;
        int K = (i < 8) ? (8 * g + i) : (16 + 8 * g + (i - 8));
        int kglob = ko * 32 + K;        // channel index (0..319)
        int nglob = nt * 16 + n;        // output column (0..1535)
        const float* W = (nglob < 512) ? Wq : (nglob < 1024 ? Wk : Wv);
        int r = nglob & 511;
        pk[idx] = (_Float16)W[r * CCH + kglob];
    } else if (idx < QKV_PACK_HALVES + O_PACK_HALVES) {
        int id2 = idx - QKV_PACK_HALVES;
        int tile = id2 >> 9, rem = id2 & 511;
        int lane = rem >> 4, i = rem & 15;
        int co = tile / KT_O, ko = tile % KT_O;
        int g = lane >> 4, n = lane & 15;
        int K = (i < 8) ? (8 * g + i) : (16 + 8 * g + (i - 8));
        int kglob = ko * 32 + K;        // inner index (0..511)
        int nglob = co * 16 + n;        // channel column (0..319)
        pk[idx] = (_Float16)Wo[nglob * INNER + kglob];
    }
}

// ---------------------------------------------------------------------------
// Kernel 1: fused LN -> QKV -> attention -> out-proj -> +bias +residual.
// One workgroup (8 wave32) handles 2 sequences (32 token rows).
// Dynamic LDS layout (bytes):
//   X0T f32 [5120 rows=(c,tok)][2 seq]  (TDM tile) 40960
//   XNf f16 A-frags  [2 rt][10 ko][512]            20480
//   QF  f16 A-frags  [2 seq][8 h][2 ko][512]       32768
//   KF  f16 B-frags  (K^T)  same shape             32768
//   VF  f16 B-frags  [2][8][4 nt][32 ln][8]        32768
//   OF  f16 A-frags  [2 rt][16 ko][512]            32768
//   PS  f16 per-wave P scratch [8][16][16]          4096
//                                           total 196608
// ---------------------------------------------------------------------------
__launch_bounds__(256, 1)
__global__ void ca_attn_fused(const float* __restrict__ hidden,
                              const float* __restrict__ ln_w,
                              const float* __restrict__ ln_b,
                              const _Float16* __restrict__ pk,
                              const float* __restrict__ bo,
                              float* __restrict__ out) {
    extern __shared__ char smem[];
    float*    X0T = (float*)smem;                      // 5120*2 f32
    _Float16* XNf = (_Float16*)(smem + 40960);         // 10240 halves
    _Float16* QF  = XNf + 10240;                       // 16384 halves
    _Float16* KF  = QF + 16384;                        // 16384 halves
    _Float16* VF  = KF + 16384;                        // 16384 halves
    _Float16* OF  = VF + 16384;                        // 16384 halves
    _Float16* PS  = OF + 16384;                        // 2048 halves

    const int tid  = threadIdx.x;
    const int wave = tid >> 5;
    const int lane = tid & 31;
    const int hi   = lane >> 4;      // lane group
    const int ln16 = lane & 15;
    const int s0   = blockIdx.x * SEQ_WG;
    const int bb0  = s0 >> 10;
    const int hw0  = s0 & 1023;      // s0 even -> both seqs share bb, hw0|hw0+1

    // ---- Stage A: TDM 2-D tensor load of x0 into LDS ----------------------
    // Tile: tile_dim0 = 2 f32 (the 2 sequences, contiguous in hw),
    //       tile_dim1 = 5120 rows = (c,tok), row stride = 1024 f32.
    // D# built per ISA 8.3/8.4; groups 2/3 zero (2-D tensor). TENSORcnt-tracked.
    if (wave == 0) {
        const float* gsrc = hidden + ((size_t)bb0 * CCH) * TTOK * HW + hw0;
        unsigned long long ga = (unsigned long long)(const void*)gsrc;
        u32x4 g0;
        g0.x = 1u;                                        // count=1, user D#
        g0.y = (unsigned)(unsigned long long)(void*)X0T;  // lds_addr (bytes)
        g0.z = (unsigned)(ga & 0xffffffffull);            // global_addr[31:0]
        g0.w = (unsigned)((ga >> 32) & 0x01ffffffull)     // global_addr[56:32]
             | 0x80000000u;                               // type=2 ("image")
        i32x8 g1;
        g1[0] = 0x00020000;            // workgroup_mask=0, data_size=2 (4 B)
        g1[1] = (int)(1024u << 16);    // tensor_dim0 = 1024 (hw extent)
        g1[2] = (int)(5120u << 16);    // tensor_dim1 = 5120 (c*t rows)
        g1[3] = (int)(2u << 16);       // tile_dim0 = 2
        g1[4] = 5120;                  // tile_dim1 = 5120
        g1[5] = 1024;                  // tensor_dim0_stride = 1024 elements
        g1[6] = 0;
        g1[7] = 0;
        i32x4 gz4 = {0, 0, 0, 0};
        i32x8 gz8 = {0, 0, 0, 0, 0, 0, 0, 0};
        // 6-arg (clang-23 / therock-10.0) form of the TDM builtin
        __builtin_amdgcn_tensor_load_to_lds(g0, g1, gz4, gz4, gz8, 0);
        __builtin_amdgcn_s_wait_tensorcnt(0);
    }
    __syncthreads();

    // ---- LayerNorm: 8 lanes per row, shuffle-reduce, scatter f16 A-frags ----
    {
        int row  = tid >> 3;           // (seq,tok)
        int part = tid & 7;
        int seq  = row >> 4, tok = row & 15;
        float sum = 0.f, sq = 0.f;
        float xv[40];
        for (int k = 0; k < 40; ++k) {
            int c = part * 40 + k;
            float v = X0T[((c * TTOK) + tok) * 2 + seq];
            xv[k] = v; sum += v; sq += v * v;
        }
        for (int mask = 1; mask < 8; mask <<= 1) {
            sum += __shfl_xor(sum, mask);
            sq  += __shfl_xor(sq,  mask);
        }
        float mu   = sum * (1.f / 320.f);
        float var  = sq * (1.f / 320.f) - mu * mu;
        float rstd = rsqrtf(var + 1e-5f);
        for (int k = 0; k < 40; ++k) {
            int   c  = part * 40 + k;
            float xn = (xv[k] - mu) * rstd * ln_w[c] + ln_b[c];
            int ko = c >> 5, kk = c & 31;
            int g = frag_g_from_kk(kk), i = frag_i_from_kk(kk);
            XNf[((seq * KT_X + ko) << 9) + ((g * 16 + tok) << 4) + i] = (_Float16)xn;
        }
    }
    __syncthreads();

    // ---- Stage B: QKV GEMM, 192 tiles of 16x16, 10 WMMA k-steps each ----
    for (int tt = wave; tt < 2 * NT_QKV; tt += 8) {
        int rt = tt & 1;          // row tile == local sequence
        int nt = tt >> 1;         // column tile in [Q|K|V]
        const _Float16* bbase = pk + (size_t)(nt * KT_X) * 512;
        __builtin_prefetch(bbase + (lane << 4), 0, 1);  // global_prefetch_b8
        v8f acc = {};
        for (int ko = 0; ko < KT_X; ++ko) {
            v16h a = *(const v16h*)(XNf + ((rt * KT_X + ko) << 9) + (lane << 4));
            v16h b = *(const v16h*)(bbase + (ko << 9) + (lane << 4));
            acc = wmma_f16(a, b, acc);
        }
        // scatter C (row tok = j + 8*hi, col = nt*16 + ln16) into frag layouts
        int cg    = nt * 16 + ln16;
        int which = cg >> 9;            // 0=Q 1=K 2=V (uniform per tile)
        int cw    = cg & 511;
        int head  = cw >> 6, d = cw & 63;
        if (which == 2) {
            // V B-frag: lane slot = hi*16 + (d&15), halves i = tok&7 = j
            // -> the 8 results are 8 contiguous halves: one ds_store_b128.
            v8h pv;
            for (int j = 0; j < 8; ++j) pv[j] = (_Float16)acc[j];
            *(v8h*)(VF + ((((rt * 8 + head) * 4 + (d >> 4)) * 32 +
                           hi * 16 + (d & 15)) << 3)) = pv;
        } else {
            int ko2 = d >> 5, kk = d & 31;
            int g = frag_g_from_kk(kk), i = frag_i_from_kk(kk);
            _Float16* dst = (which == 0) ? QF : KF;  // Q: A-frag / K^T: B-frag
            for (int j = 0; j < 8; ++j) {
                int tok = j + (hi << 3);
                dst[(((rt * 8 + head) * 2 + ko2) << 9) +
                    ((g * 16 + tok) << 4) + i] = (_Float16)acc[j];
            }
        }
    }
    __syncthreads();

    // ---- Stage C: attention, one (seq,head) per wave iteration ----
    for (int task = wave; task < SEQ_WG * HEADS; task += 8) {
        int seq = task >> 3, head = task & 7;
        const _Float16* qb = QF + (((seq * 8 + head) * 2) << 9);
        const _Float16* kb = KF + (((seq * 8 + head) * 2) << 9);
        v16h a0 = *(const v16h*)(qb + (lane << 4));
        v16h a1 = *(const v16h*)(qb + 512 + (lane << 4));
        v16h b0 = *(const v16h*)(kb + (lane << 4));
        v16h b1 = *(const v16h*)(kb + 512 + (lane << 4));
        v8f sc = {};
        sc = wmma_f16(a0, b0, sc);
        sc = wmma_f16(a1, b1, sc);      // S = Q K^T over d=64

        // row softmax: row j spans 16 lanes of one C VGPR -> xor-shuffle reduce
        float pj[8];
        for (int j = 0; j < 8; ++j) {
            float x  = sc[j] * 0.125f;  // DIM_HEAD^-0.5
            float mx = x;
            for (int mask = 1; mask < 16; mask <<= 1)
                mx = fmaxf(mx, __shfl_xor(mx, mask));
            float e  = __expf(x - mx);
            float sm = e;
            for (int mask = 1; mask < 16; mask <<= 1)
                sm += __shfl_xor(sm, mask);
            pj[j] = e / sm;
        }

        // transpose P through per-wave LDS scratch into A-frag form (K padded)
        _Float16* ps = PS + (wave << 8);
        for (int j = 0; j < 8; ++j)
            ps[((j + (hi << 3)) << 4) + ln16] = (_Float16)pj[j];
        asm volatile("" ::: "memory");  // keep store->load order; LDS in-order
        v8h pa8 = *(const v8h*)(ps + (ln16 << 4) + (hi << 3));
        v16h pa;
        for (int i = 0; i < 8; ++i) { pa[i] = pa8[i]; pa[i + 8] = (_Float16)0.f; }

        for (int ntile = 0; ntile < 4; ++ntile) {
            v8h vb8 = *(const v8h*)(VF + ((((seq * 8 + head) * 4 + ntile) * 32 +
                                           lane) << 3));
            v16h vb;
            for (int i = 0; i < 8; ++i) { vb[i] = vb8[i]; vb[i + 8] = (_Float16)0.f; }
            v8f oacc = {};
            oacc = wmma_f16(pa, vb, oacc);      // O = P V
            for (int j = 0; j < 8; ++j) {
                int tok = j + (hi << 3);
                int inr = head * 64 + ntile * 16 + ln16;
                int ko = inr >> 5, kk = inr & 31;
                int g = frag_g_from_kk(kk), i2 = frag_i_from_kk(kk);
                OF[((seq * KT_O + ko) << 9) + ((g * 16 + tok) << 4) + i2] =
                    (_Float16)oacc[j];
            }
        }
    }
    __syncthreads();

    // ---- Stage D: out-proj GEMM + bias + residual, write (b,c,t,h,w) ----
    const _Float16* wo = pk + (size_t)QKV_PACK_HALVES;
    for (int tt = wave; tt < 2 * NT_OUT; tt += 8) {
        int rt = tt & 1, co = tt >> 1;
        const _Float16* bbase = wo + (size_t)(co * KT_O) * 512;
        __builtin_prefetch(bbase + (lane << 4), 0, 1);
        v8f acc = {};
        for (int ko = 0; ko < KT_O; ++ko) {
            v16h a = *(const v16h*)(OF + ((rt * KT_O + ko) << 9) + (lane << 4));
            v16h b = *(const v16h*)(bbase + (ko << 9) + (lane << 4));
            acc = wmma_f16(a, b, acc);
        }
        int c = co * 16 + ln16;
        int s = s0 + rt;
        int bb = s >> 10, hw = s & 1023;
        float bias = bo[c];
        for (int j = 0; j < 8; ++j) {
            int tok = j + (hi << 3);
            size_t gi = (((size_t)bb * CCH + c) * TTOK + tok) * HW + hw;
            out[gi] = acc[j] + bias + hidden[gi];   // residual: L2-hot re-read
        }
    }
}

// ---------------------------------------------------------------------------
extern "C" void kernel_launch(void* const* d_in, const int* in_sizes, int n_in,
                              void* d_out, int out_size, void* d_ws, size_t ws_size,
                              hipStream_t stream) {
    (void)in_sizes; (void)n_in; (void)out_size; (void)ws_size;
    const float* hidden = (const float*)d_in[0];
    const float* ln_w   = (const float*)d_in[1];
    const float* ln_b   = (const float*)d_in[2];
    const float* Wq     = (const float*)d_in[3];
    const float* Wk     = (const float*)d_in[4];
    const float* Wv     = (const float*)d_in[5];
    const float* Wo     = (const float*)d_in[6];
    const float* bo     = (const float*)d_in[7];
    _Float16*    pk     = (_Float16*)d_ws;       // 1.31 MB packed f16 weights

    const int totalPack = QKV_PACK_HALVES + O_PACK_HALVES;
    ca_pack_weights<<<(totalPack + 255) / 256, 256, 0, stream>>>(Wq, Wk, Wv, Wo, pk);

    const size_t smem = 40960 + 2 * (size_t)(10240 + 4 * 16384 + 2048); // 196608 B
    (void)hipFuncSetAttribute((const void*)ca_attn_fused,
                              hipFuncAttributeMaxDynamicSharedMemorySize, (int)smem);
    ca_attn_fused<<<NSEQ / SEQ_WG, 256, smem, stream>>>(
        hidden, ln_w, ln_b, pk, bo, (float*)d_out);
}
